// GCN_with_linear_49692771615404
// MI455X (gfx1250) — compile-verified, matrix operand order checked by reading, not run on hardware
//
#include <hip/hip_runtime.h>
#include <hip/hip_bf16.h>
#include <math.h>

// ---------------------------------------------------------------------------
// GCN (2x GCNConv + linear + log_softmax) for MI455X (gfx1250, wave32, WMMA)
// ---------------------------------------------------------------------------

typedef __attribute__((ext_vector_type(2)))  float    v2f;
typedef __attribute__((ext_vector_type(8)))  float    v8f;
typedef __attribute__((ext_vector_type(16))) _Float16 v16h;

#define GCN_N     100000
#define GCN_HID   16
#define GCN_NCLS  40

#if defined(__has_builtin)
#if __has_builtin(__builtin_amdgcn_wmma_f32_16x16x4_f32)
#define GCN_USE_F32_WMMA 1
#endif
#endif

// ---------------------------------------------------------------------------
// Degree / normalization kernels
// ---------------------------------------------------------------------------
__global__ void gcn_deg_init(int* __restrict__ deg, int n) {
  int i = blockIdx.x * blockDim.x + threadIdx.x;
  if (i < n) deg[i] = 1;            // self loop
}

__global__ void gcn_deg_count(const int* __restrict__ col, int* __restrict__ deg, int e) {
  int i = blockIdx.x * blockDim.x + threadIdx.x;
  if (i < e) atomicAdd(&deg[col[i]], 1);
}

__global__ void gcn_dinv(float* __restrict__ dinv_buf, int n) {
  int i = blockIdx.x * blockDim.x + threadIdx.x;
  if (i < n) {
    float d = (float)((const int*)dinv_buf)[i];   // deg >= 1 always (self loop)
    dinv_buf[i] = rsqrtf(d);
  }
}

// ---------------------------------------------------------------------------
// WMMA GEMM: H[N x 16] = A[N x K] @ W[K x 16]
// One wave per 16-row tile; W staged in LDS; 8 waves (256 threads) per block.
// ---------------------------------------------------------------------------
#if defined(GCN_USE_F32_WMMA)

template <int K>
__global__ __launch_bounds__(256) void gcn_gemm(const float* __restrict__ A,
                                                const float* __restrict__ W,
                                                float* __restrict__ Hout,
                                                int nTiles) {
  __shared__ float sW[K * 16];
  for (int i = threadIdx.x; i < K * 16; i += 256) sW[i] = W[i];
  __syncthreads();

  const int wave = __builtin_amdgcn_readfirstlane((int)(threadIdx.x >> 5));
  const int lane = (int)(threadIdx.x & 31);
  const int tile = blockIdx.x * 8 + wave;
  if (tile >= nTiles) return;      // wave-uniform: EXEC stays all-ones for WMMA

  const int m    = lane & 15;       // row within tile (A) / column (B,C)
  const int koff = (lane >> 4) << 1;

  const float* arow = A + ((size_t)tile * 16 + m) * K + koff;
  v8f acc = {};

#pragma unroll 4
  for (int kk = 0; kk < K; kk += 4) {
    v2f a = *(const v2f*)(arow + kk);                 // A[m][kk+koff .. +1]
    v2f b;
    b.x = sW[(kk + koff) * 16 + m];                   // B[kk+koff  ][n]
    b.y = sW[(kk + koff + 1) * 16 + m];               // B[kk+koff+1][n]
    acc = __builtin_amdgcn_wmma_f32_16x16x4_f32(
        false, a, false, b, (short)0, acc, false, false);
  }

  // C layout: VGPR r -> (M = r | r+8, N = lane&15)
  const int mbase = (lane >> 4) << 3;
  float* orow = Hout + (size_t)tile * 16 * 16;
#pragma unroll
  for (int r = 0; r < 8; ++r)
    orow[(mbase + r) * 16 + m] = acc[r];
}

#else  // fallback: codegen-confirmed f16 WMMA (convert on load, fp32 accum)

template <int K>
__global__ __launch_bounds__(256) void gcn_gemm(const float* __restrict__ A,
                                                const float* __restrict__ W,
                                                float* __restrict__ Hout,
                                                int nTiles) {
  constexpr int KP = (K + 31) & ~31;                  // pad K to multiple of 32
  __shared__ _Float16 sW[KP * 16];
  for (int i = threadIdx.x; i < KP * 16; i += 256)
    sW[i] = (i < K * 16) ? (_Float16)W[i] : (_Float16)0.0f;
  __syncthreads();

  const int wave = __builtin_amdgcn_readfirstlane((int)(threadIdx.x >> 5));
  const int lane = (int)(threadIdx.x & 31);
  const int tile = blockIdx.x * 8 + wave;
  if (tile >= nTiles) return;

  const int m   = lane & 15;
  const int sel = (lane >> 4) << 3;                   // 0 or 8

  const float* arow = A + ((size_t)tile * 16 + m) * K;
  v8f acc = {};

  for (int kk = 0; kk < KP; kk += 32) {
    v16h a, b;
#pragma unroll
    for (int i = 0; i < 8; ++i) {
      int k0 = kk + sel + i;                          // halves 0..7  : K 0..7 / 8..15
      int k1 = kk + 16 + sel + i;                     // halves 8..15 : K 16..23 / 24..31
      a[i]     = (k0 < K) ? (_Float16)arow[k0] : (_Float16)0.0f;
      a[i + 8] = (k1 < K) ? (_Float16)arow[k1] : (_Float16)0.0f;
      b[i]     = sW[k0 * 16 + m];
      b[i + 8] = sW[k1 * 16 + m];
    }
    acc = __builtin_amdgcn_wmma_f32_16x16x32_f16(
        false, a, false, b, (short)0, acc, false, false);
  }

  const int mbase = (lane >> 4) << 3;
  float* orow = Hout + (size_t)tile * 16 * 16;
#pragma unroll
  for (int r = 0; r < 8; ++r)
    orow[(mbase + r) * 16 + m] = acc[r];
}

#endif  // GCN_USE_F32_WMMA

// ---------------------------------------------------------------------------
// Aggregation: agg = D^{-1/2} A_hat D^{-1/2} h   (A_hat = A + I)
// ---------------------------------------------------------------------------
// self loop: agg[i][:] = h[i][:] * dinv[i]^2     (one thread per float4)
__global__ void gcn_self_init(const float* __restrict__ h,
                              const float* __restrict__ dinv,
                              float* __restrict__ agg, int n4) {
  int i = blockIdx.x * blockDim.x + threadIdx.x;
  if (i >= n4) return;
  int node = i >> 2;
  float d = dinv[node];
  float s = d * d;
  float4 hv = ((const float4*)h)[i];
  float4 o  = make_float4(hv.x * s, hv.y * s, hv.z * s, hv.w * s);
  ((float4*)agg)[i] = o;
}

// edges: 4 threads per edge, each handles 4 of the 16 features (L2-resident atomics)
__global__ void gcn_edge_agg(const int* __restrict__ row, const int* __restrict__ col,
                             const float* __restrict__ dinv, const float* __restrict__ h,
                             float* __restrict__ agg, int nEdges) {
  int t = blockIdx.x * blockDim.x + threadIdx.x;
  int e = t >> 2;
  if (e >= nEdges) return;
  int q = t & 3;
  int r = row[e];
  int c = col[e];
  float nrm = dinv[r] * dinv[c];
  float4 hv = ((const float4*)(h + (size_t)r * 16))[q];
  float* dst = agg + (size_t)c * 16 + q * 4;
  atomicAdd(dst + 0, hv.x * nrm);
  atomicAdd(dst + 1, hv.y * nrm);
  atomicAdd(dst + 2, hv.z * nrm);
  atomicAdd(dst + 3, hv.w * nrm);
}

__global__ void gcn_bias_relu(const float* __restrict__ agg,
                              const float* __restrict__ bias,
                              float* __restrict__ out, int n16) {
  int i = blockIdx.x * blockDim.x + threadIdx.x;
  if (i >= n16) return;
  float v = agg[i] + bias[i & 15];
  out[i] = fmaxf(v, 0.0f);
}

// ---------------------------------------------------------------------------
// Head: logits = h @ Wl + bl ; out = log_softmax(logits)
// ---------------------------------------------------------------------------
__global__ __launch_bounds__(256) void gcn_head(const float* __restrict__ h,
                                                const float* __restrict__ Wl,
                                                const float* __restrict__ bl,
                                                float* __restrict__ out, int n) {
  __shared__ float sW[GCN_HID * GCN_NCLS];
  __shared__ float sB[GCN_NCLS];
  for (int i = threadIdx.x; i < GCN_HID * GCN_NCLS; i += 256) sW[i] = Wl[i];
  if (threadIdx.x < GCN_NCLS) sB[threadIdx.x] = bl[threadIdx.x];
  __syncthreads();

  int i = blockIdx.x * 256 + threadIdx.x;
  if (i >= n) return;

  float hv[GCN_HID];
  const float4* hp = (const float4*)(h + (size_t)i * GCN_HID);
#pragma unroll
  for (int q = 0; q < 4; ++q) {
    float4 v = hp[q];
    hv[q * 4 + 0] = v.x; hv[q * 4 + 1] = v.y;
    hv[q * 4 + 2] = v.z; hv[q * 4 + 3] = v.w;
  }

  float lg[GCN_NCLS];
  float mx = -INFINITY;
#pragma unroll
  for (int c = 0; c < GCN_NCLS; ++c) {
    float s = sB[c];
#pragma unroll
    for (int j = 0; j < GCN_HID; ++j)
      s = fmaf(hv[j], sW[j * GCN_NCLS + c], s);
    lg[c] = s;
    mx = fmaxf(mx, s);
  }
  float sum = 0.0f;
#pragma unroll
  for (int c = 0; c < GCN_NCLS; ++c) sum += expf(lg[c] - mx);
  float lse = mx + logf(sum);

  float* orow = out + (size_t)i * GCN_NCLS;
#pragma unroll
  for (int c = 0; c < GCN_NCLS; ++c) orow[c] = lg[c] - lse;
}

// ---------------------------------------------------------------------------
// Launch
// ---------------------------------------------------------------------------
extern "C" void kernel_launch(void* const* d_in, const int* in_sizes, int n_in,
                              void* d_out, int out_size, void* d_ws, size_t ws_size,
                              hipStream_t stream) {
  const float* x  = (const float*)d_in[0];
  const int*   ei = (const int*)  d_in[1];
  const float* W1 = (const float*)d_in[2];
  const float* b1 = (const float*)d_in[3];
  const float* W2 = (const float*)d_in[4];
  const float* b2 = (const float*)d_in[5];
  const float* Wl = (const float*)d_in[6];
  const float* bl = (const float*)d_in[7];

  const int N = in_sizes[0] / 512;     // 100000
  const int E = in_sizes[1] / 2;       // 3200000
  const int* row = ei;                 // sources
  const int* col = ei + E;             // targets

  float* ws   = (float*)d_ws;
  float* dinv = ws;                           // N floats (also int deg scratch)
  float* bufA = ws + (size_t)N;               // N*16
  float* bufB = bufA + (size_t)N * 16;        // N*16
  float* bufC = bufB + (size_t)N * 16;        // N*16

  const int nTiles = N / 16;                  // 6250
  const int TPB    = 256;
  const int gN   = (N + TPB - 1) / TPB;
  const int gE   = (E + TPB - 1) / TPB;
  const int gN4  = (N * 4 + TPB - 1) / TPB;
  const int gN16 = (N * 16 + TPB - 1) / TPB;
  const int gE4  = (int)(((size_t)E * 4 + TPB - 1) / TPB);
  const int gGemm = (nTiles + 7) / 8;

  // normalization
  gcn_deg_init <<<gN, TPB, 0, stream>>>((int*)dinv, N);
  gcn_deg_count<<<gE, TPB, 0, stream>>>(col, (int*)dinv, E);
  gcn_dinv     <<<gN, TPB, 0, stream>>>(dinv, N);

  // layer 1
  gcn_gemm<512><<<gGemm, TPB, 0, stream>>>(x, W1, bufA, nTiles);
  gcn_self_init<<<gN4,  TPB, 0, stream>>>(bufA, dinv, bufB, N * 4);
  gcn_edge_agg <<<gE4,  TPB, 0, stream>>>(row, col, dinv, bufA, bufB, E);
  gcn_bias_relu<<<gN16, TPB, 0, stream>>>(bufB, b1, bufA, N * 16);

  // layer 2
  gcn_gemm<16> <<<gGemm, TPB, 0, stream>>>(bufA, W2, bufB, nTiles);
  gcn_self_init<<<gN4,  TPB, 0, stream>>>(bufB, dinv, bufC, N * 4);
  gcn_edge_agg <<<gE4,  TPB, 0, stream>>>(row, col, dinv, bufB, bufC, E);
  gcn_bias_relu<<<gN16, TPB, 0, stream>>>(bufC, b2, bufA, N * 16);

  // classifier head + log_softmax
  gcn_head<<<gN, TPB, 0, stream>>>(bufA, Wl, bl, (float*)d_out, N);
}
